// rlf_40261023433259
// MI455X (gfx1250) — matrix operand order repaired and checked by reading, not compiled, not run on hardware
//
#include <hip/hip_runtime.h>
#include <hip/hip_bf16.h>

// ---------------------------------------------------------------------------
// Hypernetwork forward for MI455X (gfx1250, wave32).
//
//   1. CNN encoder run once per UNIQUE map (64), not per sample (4096).
//   2. fc1 / layernorm / bottleneck / hypernet heads as WMMA f16 GEMMs
//      (M=64 rows, f32 accumulation, v_wmma_f32_16x16x32_f16) with the
//      shared B tile staged through LDS (coalesced f32 loads -> f16,
//      transposed, conflict-free ds_load_b128 fragment reads).
//   3. Per-sample tiny MLP (unique weights per sample -> VALU/LDS GEMV),
//      then softmax over 8 outputs.
// ---------------------------------------------------------------------------

#define ALPHA  0.2f
#define LN_EPS 1e-3f
#define TH_W   18184   // theta row width: 512+128+16384+128+1024+8

typedef __attribute__((ext_vector_type(16))) _Float16 v16h;
typedef __attribute__((ext_vector_type(8)))  _Float16 v8h;
typedef __attribute__((ext_vector_type(8)))  float    v8f;

__device__ __forceinline__ float lrelu(float x) { return x >= 0.f ? x : ALPHA * x; }

// ---------------------------------------------------------------------------
// Generic NHWC conv, one block per map, SAME padding via bounds check.
// ---------------------------------------------------------------------------
__global__ void k_conv(const float* __restrict__ in, float* __restrict__ out,
                       const float* __restrict__ w, const float* __restrict__ bias,
                       int Hin, int Cin, int Hout, int Cout,
                       int ksz, int stride, int pad,
                       int inMapStride, int outMapStride, int act)
{
    const int map = blockIdx.x;
    in  += (size_t)map * inMapStride;
    out += (size_t)map * outMapStride;
    const int total = Hout * Hout * Cout;
    for (int idx = threadIdx.x; idx < total; idx += blockDim.x) {
        const int co = idx % Cout;
        const int x  = (idx / Cout) % Hout;
        const int y  = idx / (Cout * Hout);
        float acc = bias[co];
        for (int ky = 0; ky < ksz; ++ky) {
            const int iy = y * stride + ky - pad;
            if (iy < 0 || iy >= Hin) continue;
            for (int kx = 0; kx < ksz; ++kx) {
                const int ix = x * stride + kx - pad;
                if (ix < 0 || ix >= Hin) continue;
                const float* ip = in + (size_t)(iy * Hin + ix) * Cin;
                const float* wp = w + (size_t)((ky * ksz + kx) * Cin) * Cout + co;
                for (int ci = 0; ci < Cin; ++ci)
                    acc = __builtin_fmaf(ip[ci], wp[(size_t)ci * Cout], acc);
            }
        }
        if (act) acc = lrelu(acc);
        out[idx] = acc;
    }
}

// ---------------------------------------------------------------------------
// Residual block: out = base + lrelu(conv3x3_1(base)) + lrelu(conv3x3_2(base))
// Base tile (<= 4096 floats) staged in LDS.
// ---------------------------------------------------------------------------
__global__ void k_res(const float* __restrict__ base, float* __restrict__ out,
                      const float* __restrict__ w1, const float* __restrict__ b1,
                      const float* __restrict__ w2, const float* __restrict__ b2,
                      int H, int C, int mapStride)
{
    __shared__ float sb[4096];
    const int map = blockIdx.x;
    base += (size_t)map * mapStride;
    out  += (size_t)map * mapStride;
    const int total = H * H * C;
    for (int i = threadIdx.x; i < total; i += blockDim.x) sb[i] = base[i];
    __syncthreads();
    for (int idx = threadIdx.x; idx < total; idx += blockDim.x) {
        const int co = idx % C;
        const int x  = (idx / C) % H;
        const int y  = idx / (C * H);
        float a1 = b1[co], a2 = b2[co];
        for (int ky = 0; ky < 3; ++ky) {
            const int iy = y + ky - 1;
            if (iy < 0 || iy >= H) continue;
            for (int kx = 0; kx < 3; ++kx) {
                const int ix = x + kx - 1;
                if (ix < 0 || ix >= H) continue;
                const float* sp  = sb + (size_t)(iy * H + ix) * C;
                const float* w1p = w1 + (size_t)((ky * 3 + kx) * C) * C + co;
                const float* w2p = w2 + (size_t)((ky * 3 + kx) * C) * C + co;
                for (int ci = 0; ci < C; ++ci) {
                    a1 = __builtin_fmaf(sp[ci], w1p[(size_t)ci * C], a1);
                    a2 = __builtin_fmaf(sp[ci], w2p[(size_t)ci * C], a2);
                }
            }
        }
        out[idx] = sb[idx] + lrelu(a1) + lrelu(a2);
    }
}

// ---------------------------------------------------------------------------
// WMMA GEMM: C[64 x N] = act(A[64 x K] @ B[K x N] + bias), f16 in / f32 acc.
// blockDim = 128 (4 waves); wave w owns row-tile m0 = 16*w; blockIdx.x = N tile.
// Requirements: K % 32 == 0, N % 16 == 0 (true for all launches below).
//
// B tile (32 x 16 fp32) is staged per block: 128 threads do coalesced float4
// row loads, convert to f16 and store TRANSPOSED into LDS as sB[n][k] with a
// row stride of 40 halves (80B). Fragment reads are then two 16B-aligned v8h
// chunks per lane at byte offsets lh*80 + hg*16 + {0,32} -> ds_load_b128, and the
// 20-word row stride puts the 16 lanes of each half-group on disjoint bank
// sets (conflict-free).
// ---------------------------------------------------------------------------
__global__ void k_gemm_wmma(const float* __restrict__ A, const float* __restrict__ B,
                            const float* __restrict__ bias, float* __restrict__ C,
                            int N, int K, int ldc, int col_off, int act)
{
    __shared__ __align__(16) _Float16 sB[16 * 40];  // [n][k], k-stride 40 halves

    const int tid  = threadIdx.x;
    const int lane = tid & 31;
    const int wave = tid >> 5;
    const int m0 = wave * 16;
    const int n0 = blockIdx.x * 16;
    const int lh = lane & 15;   // lane within 16-lane half
    const int hg = lane >> 4;   // half-group (0 or 1)

    // B staging role: row br (k within tile), 4-wide column group bc (n)
    const int br = tid >> 2;          // 0..31
    const int bc = (tid & 3) * 4;     // 0,4,8,12

    v8f acc = {};
    const int m = m0 + lh;            // A row handled by this lane

    for (int kb = 0; kb < K; kb += 32) {
        // ---- coalesced B tile load (shared by all 4 waves) ----
        const float4 bv = *(const float4*)(B + (size_t)(kb + br) * N + n0 + bc);
        if (kb + 32 < K)
            __builtin_prefetch(B + (size_t)(kb + 32 + br) * N + n0 + bc, 0, 1);

        __syncthreads();  // previous iteration's fragment reads done
        sB[(bc + 0) * 40 + br] = (_Float16)bv.x;
        sB[(bc + 1) * 40 + br] = (_Float16)bv.y;
        sB[(bc + 2) * 40 + br] = (_Float16)bv.z;
        sB[(bc + 3) * 40 + br] = (_Float16)bv.w;
        __syncthreads();

        // ---- A fragment (16x32 f16), straight from global (b128-vectorizes)
        // VGPR i<4 -> K = 2i + 8*hg ; i>=4 -> K = 16 + 2(i-4) + 8*hg
        v16h af;
        const float* Ap = A + (size_t)m * K + kb + hg * 8;
        #pragma unroll
        for (int i = 0; i < 8; ++i) {
            const int kk = (i < 4) ? (2 * i) : (16 + 2 * (i - 4));
            af[2 * i]     = (_Float16)Ap[kk];
            af[2 * i + 1] = (_Float16)Ap[kk + 1];
        }

        // ---- B fragment from LDS: half j<8 -> K = kb + 8*hg + j, j>=8 -> K = kb + 16 + 8*hg + (j-8), lane -> N
        const _Float16* bp = &sB[lh * 40 + hg * 8];
        const v8h b0 = *(const v8h*)bp;
        const v8h b1 = *(const v8h*)(bp + 16);
        v16h bf;
        #pragma unroll
        for (int j = 0; j < 8; ++j) { bf[j] = b0[j]; bf[8 + j] = b1[j]; }

        // EXEC is all-ones here (no divergent exits anywhere in this kernel).
        acc = __builtin_amdgcn_wmma_f32_16x16x32_f16(
            false, af, false, bf, (short)0, acc, false, false);
    }

    const int nn = n0 + lh;
    const float bvs = bias[nn];
    #pragma unroll
    for (int r = 0; r < 8; ++r) {
        const int mm = m0 + r + hg * 8;  // C/D layout: VGPR r -> M=r (+8 upper half)
        float v = acc[r] + bvs;
        if (act) v = lrelu(v);
        C[(size_t)mm * ldc + col_off + nn] = v;
    }
}

// ---------------------------------------------------------------------------
// Tiny head (N=8, below a WMMA tile): theta[:, col..col+7] = z2 @ w + b
// ---------------------------------------------------------------------------
__global__ void k_head8(const float* __restrict__ z2, const float* __restrict__ w,
                        const float* __restrict__ b, float* __restrict__ theta,
                        int col_off)
{
    const int mrow = blockIdx.x;       // 64 maps
    const int t = threadIdx.x;         // 32 threads, lanes 0..7 active
    if (t < 8) {
        float acc = b[t];
        const float* zp = z2 + (size_t)mrow * 128;
        for (int k = 0; k < 128; ++k)
            acc = __builtin_fmaf(zp[k], w[k * 8 + t], acc);
        theta[(size_t)mrow * TH_W + col_off + t] = acc;
    }
}

// ---------------------------------------------------------------------------
// LayerNorm over 256 features, one block (256 threads) per map row, in place.
// ---------------------------------------------------------------------------
__global__ void k_layernorm(float* __restrict__ z, const float* __restrict__ g,
                            const float* __restrict__ b)
{
    __shared__ float red[256];
    float* zp = z + (size_t)blockIdx.x * 256;
    const int t = threadIdx.x;
    const float v = zp[t];
    red[t] = v; __syncthreads();
    for (int s = 128; s > 0; s >>= 1) { if (t < s) red[t] += red[t + s]; __syncthreads(); }
    const float mu = red[0] / 256.f;
    __syncthreads();
    const float d = v - mu;
    red[t] = d * d; __syncthreads();
    for (int s = 128; s > 0; s >>= 1) { if (t < s) red[t] += red[t + s]; __syncthreads(); }
    const float var = red[0] / 256.f;
    zp[t] = d * rsqrtf(var + LN_EPS) * g[t] + b[t];
}

// ---------------------------------------------------------------------------
// Per-sample MLP. Weights are unique per sample (gathered from theta via
// indices) so these are GEMVs -> VALU/LDS, fed from L2 (4MB unique theta).
// ---------------------------------------------------------------------------
__global__ void k_mlp0(const float* __restrict__ states, const float* __restrict__ theta,
                       const int* __restrict__ idx, float* __restrict__ y1)
{
    const int bsmp = blockIdx.x;
    const float* th = theta + (size_t)idx[bsmp] * TH_W;   // W0 @0, b0 @512
    const int j = threadIdx.x;                            // 128 threads
    const float* s = states + (size_t)bsmp * 4;
    float acc = th[512 + j];
    #pragma unroll
    for (int k = 0; k < 4; ++k) acc = __builtin_fmaf(s[k], th[k * 128 + j], acc);
    y1[(size_t)bsmp * 128 + j] = lrelu(acc);
}

__global__ void k_mlp1(const float* __restrict__ y1, const float* __restrict__ theta,
                       const int* __restrict__ idx, float* __restrict__ y2)
{
    __shared__ float a[128];
    const int bsmp = blockIdx.x;
    const float* th = theta + (size_t)idx[bsmp] * TH_W + 640;  // W1 @640, b1 @17024
    const int j = threadIdx.x;                                 // 128 threads
    a[j] = y1[(size_t)bsmp * 128 + j];
    __syncthreads();
    float acc = th[16384 + j];
    #pragma unroll 4
    for (int k = 0; k < 128; ++k)
        acc = __builtin_fmaf(a[k], th[(size_t)k * 128 + j], acc);
    y2[(size_t)bsmp * 128 + j] = lrelu(acc);
}

__global__ void k_mlp2_softmax(const float* __restrict__ y2, const float* __restrict__ theta,
                               const int* __restrict__ idx, float* __restrict__ out)
{
    __shared__ float lg[8];
    const int bsmp = blockIdx.x;
    const float* th = theta + (size_t)idx[bsmp] * TH_W + 17152;  // W2 @17152, b2 @18176
    const int t = threadIdx.x;                                   // 32 threads (one wave)
    if (t < 8) {
        float acc = th[1024 + t];
        const float* yp = y2 + (size_t)bsmp * 128;
        for (int k = 0; k < 128; ++k)
            acc = __builtin_fmaf(yp[k], th[k * 8 + t], acc);
        lg[t] = acc;
    }
    __syncthreads();
    if (t < 8) {
        float mx = lg[0];
        #pragma unroll
        for (int o = 1; o < 8; ++o) mx = fmaxf(mx, lg[o]);
        float sum = 0.f;
        #pragma unroll
        for (int o = 0; o < 8; ++o) sum += __expf(lg[o] - mx);
        out[(size_t)bsmp * 8 + t] = __expf(lg[t] - mx) / sum;
    }
}

// ---------------------------------------------------------------------------
// Launch. Inputs flattened in setup_inputs() dict order (nested dicts
// depth-first in insertion order):
//  0 states  1 maps  2..33 enc(c0w..bnb)  34..45 hyp(fW0w..fb2b)  46 indices
// ---------------------------------------------------------------------------
extern "C" void kernel_launch(void* const* d_in, const int* in_sizes, int n_in,
                              void* d_out, int out_size, void* d_ws, size_t ws_size,
                              hipStream_t stream)
{
    (void)in_sizes; (void)n_in; (void)out_size; (void)ws_size;

    const float* states = (const float*)d_in[0];
    const float* maps   = (const float*)d_in[1];
    const float* c0w  = (const float*)d_in[2];  const float* c0b  = (const float*)d_in[3];
    const float* c1w  = (const float*)d_in[4];  const float* c1b  = (const float*)d_in[5];
    const float* c11w = (const float*)d_in[6];  const float* c11b = (const float*)d_in[7];
    const float* c12w = (const float*)d_in[8];  const float* c12b = (const float*)d_in[9];
    const float* c2w  = (const float*)d_in[10]; const float* c2b  = (const float*)d_in[11];
    const float* c21w = (const float*)d_in[12]; const float* c21b = (const float*)d_in[13];
    const float* c22w = (const float*)d_in[14]; const float* c22b = (const float*)d_in[15];
    const float* c3w  = (const float*)d_in[16]; const float* c3b  = (const float*)d_in[17];
    const float* c31w = (const float*)d_in[18]; const float* c31b = (const float*)d_in[19];
    const float* c32w = (const float*)d_in[20]; const float* c32b = (const float*)d_in[21];
    const float* c4w  = (const float*)d_in[22]; const float* c4b  = (const float*)d_in[23];
    const float* c41w = (const float*)d_in[24]; const float* c41b = (const float*)d_in[25];
    const float* c42w = (const float*)d_in[26]; const float* c42b = (const float*)d_in[27];
    const float* fc1w = (const float*)d_in[28]; const float* fc1b = (const float*)d_in[29];
    const float* ln_g = (const float*)d_in[30]; const float* ln_b = (const float*)d_in[31];
    const float* bnw  = (const float*)d_in[32]; const float* bnb  = (const float*)d_in[33];
    const float* fW0w = (const float*)d_in[34]; const float* fW0b = (const float*)d_in[35];
    const float* fb0w = (const float*)d_in[36]; const float* fb0b = (const float*)d_in[37];
    const float* fW1w = (const float*)d_in[38]; const float* fW1b = (const float*)d_in[39];
    const float* fb1w = (const float*)d_in[40]; const float* fb1b = (const float*)d_in[41];
    const float* fW2w = (const float*)d_in[42]; const float* fW2b = (const float*)d_in[43];
    const float* fb2w = (const float*)d_in[44]; const float* fb2b = (const float*)d_in[45];
    const int* indices = (const int*)d_in[46];

    // Workspace layout (floats). Total ~3.59M floats (~14.4 MB).
    float* ws    = (float*)d_ws;
    float* zfeat = ws + 0;        // 64 x 512
    float* z1    = ws + 32768;    // 64 x 256
    float* z2    = ws + 49152;    // 64 x 128
    float* theta = ws + 57344;    // 64 x 18184
    float* y1    = ws + 1221120;  // 4096 x 128
    float* y2    = ws + 1745408;  // 4096 x 128
    float* bufA  = ws + 2269696;  // 64 x 16384 (conv ping)
    float* bufB  = ws + 3318272;  // 64 x 4096  (conv pong)

    // ---- Encoder: 64 unique maps only ----
    k_conv<<<64, 256, 0, stream>>>(maps, bufA, c0w, c0b, 32, 1, 32, 16, 3, 1, 1, 1024, 16384, 0);
    k_conv<<<64, 256, 0, stream>>>(bufA, bufB, c1w, c1b, 32, 16, 16, 16, 5, 2, 1, 16384, 4096, 1);
    k_res <<<64, 256, 0, stream>>>(bufB, bufA, c11w, c11b, c12w, c12b, 16, 16, 4096);
    k_conv<<<64, 256, 0, stream>>>(bufA, bufB, c2w, c2b, 16, 16, 8, 32, 5, 2, 1, 4096, 2048, 1);
    k_res <<<64, 256, 0, stream>>>(bufB, bufA, c21w, c21b, c22w, c22b, 8, 32, 2048);
    k_conv<<<64, 256, 0, stream>>>(bufA, bufB, c3w, c3b, 8, 32, 4, 64, 5, 2, 1, 2048, 1024, 1);
    k_res <<<64, 256, 0, stream>>>(bufB, bufA, c31w, c31b, c32w, c32b, 4, 64, 1024);
    k_conv<<<64, 256, 0, stream>>>(bufA, bufB, c4w, c4b, 4, 64, 2, 128, 5, 2, 1, 1024, 512, 1);
    k_res <<<64, 256, 0, stream>>>(bufB, zfeat, c41w, c41b, c42w, c42b, 2, 128, 512);

    // ---- Dense path on WMMA (M = 64 rows; N % 16 == 0, K % 32 == 0) ----
    k_gemm_wmma<<<16,  128, 0, stream>>>(zfeat, fc1w, fc1b, z1, 256, 512, 256, 0, 1);
    k_layernorm<<<64, 256, 0, stream>>>(z1, ln_g, ln_b);
    k_gemm_wmma<<<8,   128, 0, stream>>>(z1, bnw, bnb, z2, 128, 256, 128, 0, 0);
    // Hypernetwork heads -> theta[64 x 18184]
    k_gemm_wmma<<<32,   128, 0, stream>>>(z2, fW0w, fW0b, theta, 512,   128, TH_W, 0,     0);
    k_gemm_wmma<<<8,    128, 0, stream>>>(z2, fb0w, fb0b, theta, 128,   128, TH_W, 512,   0);
    k_gemm_wmma<<<1024, 128, 0, stream>>>(z2, fW1w, fW1b, theta, 16384, 128, TH_W, 640,   0);
    k_gemm_wmma<<<8,    128, 0, stream>>>(z2, fb1w, fb1b, theta, 128,   128, TH_W, 17024, 0);
    k_gemm_wmma<<<64,   128, 0, stream>>>(z2, fW2w, fW2b, theta, 1024,  128, TH_W, 17152, 0);
    k_head8   <<<64,    32,  0, stream>>>(z2, fb2w, fb2b, theta, 18176);

    // ---- Per-sample MLP + softmax ----
    k_mlp0<<<4096, 128, 0, stream>>>(states, theta, indices, y1);
    k_mlp1<<<4096, 128, 0, stream>>>(y1, theta, indices, y2);
    k_mlp2_softmax<<<4096, 32, 0, stream>>>(y2, theta, indices, (float*)d_out);
}
